// BERTClassifier_33088428049021
// MI455X (gfx1250) — matrix operand order repaired and checked by reading, not compiled, not run on hardware
//
#include <hip/hip_runtime.h>
#include <hip/hip_bf16.h>

// ---------------- problem constants (from reference) ----------------
#define V_  30522
#define E_  768
#define F_  3072
#define L_  4
#define S_  512
#define B_  32
#define C_  2
#define M_  (S_ * B_)   // 16384 tokens

typedef __attribute__((ext_vector_type(16))) _Float16 v16h;
typedef __attribute__((ext_vector_type(8)))  _Float16 v8h;
typedef __attribute__((ext_vector_type(8)))  float    v8f;
typedef __attribute__((ext_vector_type(4)))  int      v4i;

__device__ __forceinline__ v8h ld8(const _Float16* p) { return *(const v8h*)p; }
__device__ __forceinline__ v16h cat8(v8h lo, v8h hi) {
    return __builtin_shufflevector(lo, hi, 0, 1, 2, 3, 4, 5, 6, 7,
                                   8, 9, 10, 11, 12, 13, 14, 15);
}

// ---------------- CDNA5 async global->LDS copy helpers ----------------
__device__ __forceinline__ void copy16_g2l(const _Float16* g, _Float16* l) {
#if __has_builtin(__builtin_amdgcn_global_load_async_to_lds_b128)
    __builtin_amdgcn_global_load_async_to_lds_b128((v4i*)g, (v4i*)l, 0, 0);
#else
    *(int4*)l = *(const int4*)g;
#endif
}
__device__ __forceinline__ void wait_async_le3() {
#if __has_builtin(__builtin_amdgcn_global_load_async_to_lds_b128)
#if __has_builtin(__builtin_amdgcn_s_wait_asynccnt)
    __builtin_amdgcn_s_wait_asynccnt(3);
#else
    asm volatile("s_wait_asynccnt 3" ::: "memory");
#endif
#endif
}
__device__ __forceinline__ void wait_async_le0() {
#if __has_builtin(__builtin_amdgcn_global_load_async_to_lds_b128)
#if __has_builtin(__builtin_amdgcn_s_wait_asynccnt)
    __builtin_amdgcn_s_wait_asynccnt(0);
#else
    asm volatile("s_wait_asynccnt 0" ::: "memory");
#endif
#endif
}

// ---------------- f32 -> f16 transpose-convert: dst[N,K] = (f16)src[K,N] ----------------
// One matrix per blockIdx.z (layer), threads along K (coalesced dst writes).
__global__ __launch_bounds__(256)
void cvt_t_kernel(const float* __restrict__ src, _Float16* __restrict__ dst,
                  int Kdim, int Ndim) {
    const size_t mat = (size_t)Kdim * Ndim * blockIdx.z;
    const int n = blockIdx.y;
    const int k = blockIdx.x * 256 + threadIdx.x;
    dst[mat + (size_t)n * Kdim + k] = (_Float16)src[mat + (size_t)k * Ndim + n];
}

// ---------------- embedding gather: x = tok_emb[ids] + pos_emb[s] ----------------
__global__ __launch_bounds__(256)
void embed_kernel(const int* __restrict__ ids, const float* __restrict__ tok,
                  const float* __restrict__ pos, float* __restrict__ X,
                  _Float16* __restrict__ Xh) {
    const int t   = blockIdx.x;  // token row = s*B + b
    const int tid = threadIdx.x;
    const int s   = t / B_;
    const int id  = ids[t];
#pragma unroll
    for (int i = 0; i < 3; ++i) {
        const int e = tid + i * 256;
        const float v = tok[(size_t)id * E_ + e] + pos[(size_t)s * E_ + e];
        X [(size_t)t * E_ + e] = v;
        Xh[(size_t)t * E_ + e] = (_Float16)v;
    }
}

// ---------------- WMMA tiled GEMM: C[M,N] = act(A[M,K] @ B[K,N] + bias) ----------------
// B is supplied TRANSPOSED (Bt[N,K]) so both LDS tiles stage with contiguous b128
// async copies and both fragments are contiguous LDS vector loads.
// 256 threads = 8 waves; block tile 128x64; wave tile 16x64; K step 32; 2 LDS buffers.
// Last K-step peeled so the steady-state loop has no tail branch. All four B
// fragments are loaded into distinct values before the 4 back-to-back WMMAs so
// the ds loads pipeline instead of serializing against each WMMA.
#define TM 128
#define TN 64
#define TK 32

__global__ __launch_bounds__(256)
void gemm_f16_kernel(const _Float16* __restrict__ A, const _Float16* __restrict__ Bt,
                     const float* __restrict__ bias,
                     float* __restrict__ Cf, _Float16* __restrict__ Ch,
                     int Ndim, int Kdim, int relu) {
    __shared__ __align__(16) _Float16 sA [2][TM * TK];  // 2 x 8 KB
    __shared__ __align__(16) _Float16 sBT[2][TN * TK];  // 2 x 4 KB

    const int tid  = threadIdx.x;
    const int lane = tid & 31;
    const int wave = tid >> 5;       // 0..7 -> 16-row sub-tile
    const int half = lane >> 4;
    const int l16  = lane & 15;

    const int bm = blockIdx.y * TM;
    const int bn = blockIdx.x * TN;

    v8f acc[4] = {{}, {}, {}, {}};

    // staging slots: A = 128x32 (two 16B chunks/thread), BT = 64x32 (one chunk)
    const int ac0 = tid * 2, ac1 = tid * 2 + 1;       // A chunk ids 0..511
    const int ar0 = ac0 >> 2, aco0 = (ac0 & 3) * 8;
    const int ar1 = ac1 >> 2, aco1 = (ac1 & 3) * 8;
    const int br  = tid >> 2,  bco = (tid & 3) * 8;   // BT chunk ids 0..255

    const int nsteps = Kdim / TK;    // >= 2 always (24 or 96)

    auto stage = [&](int buf, int k0) {
        copy16_g2l(A  + (size_t)(bm + ar0) * Kdim + k0 + aco0, &sA [buf][ar0 * TK + aco0]);
        copy16_g2l(A  + (size_t)(bm + ar1) * Kdim + k0 + aco1, &sA [buf][ar1 * TK + aco1]);
        copy16_g2l(Bt + (size_t)(bn + br ) * Kdim + k0 + bco,  &sBT[buf][br  * TK + bco ]);
    };
    auto compute = [&](int buf) {
        // A fragment (16x32 f16, k = 16*half + s, same order as B): contiguous 16B loads
        const _Float16* ap = &sA[buf][(wave * 16 + l16) * TK + half * 16];
        const v16h a = cat8(ld8(ap), ld8(ap + 8));
        // load ALL B fragments first (distinct registers -> ds loads pipeline)
        v16h b[4];
#pragma unroll
        for (int t = 0; t < 4; ++t) {
            // B fragment col n = t*16+l16: Bt[n, k0+half*16 .. +15] contiguous
            const _Float16* bp = &sBT[buf][(t * 16 + l16) * TK + half * 16];
            b[t] = cat8(ld8(bp), ld8(bp + 8));
        }
        // then 4 back-to-back WMMAs feeding the XDL pipe
#pragma unroll
        for (int t = 0; t < 4; ++t)
            acc[t] = __builtin_amdgcn_wmma_f32_16x16x32_f16(false, a, false, b[t],
                                                            (short)0, acc[t], false, false);
    };

    stage(0, 0);
    int st = 0;
    for (; st < nsteps - 1; ++st) {
        const int buf = st & 1;
        stage(buf ^ 1, (st + 1) * TK);
        wait_async_le3();   // the 3 loads of `buf` (issued one step earlier) are done
        __syncthreads();
        compute(buf);
        __syncthreads();
    }
    // peeled final step: no further prefetch, just drain
    wait_async_le0();
    __syncthreads();
    compute(st & 1);

    // write back: D vgpr r -> row = r + 8*half, lane -> column
    const int rowb = bm + wave * 16 + 8 * half;
    float bia[4];
#pragma unroll
    for (int t = 0; t < 4; ++t) bia[t] = bias[bn + t * 16 + l16];

    if (relu) {
#pragma unroll
        for (int t = 0; t < 4; ++t)
#pragma unroll
            for (int r = 0; r < 8; ++r) acc[t][r] = fmaxf(acc[t][r] + bia[t], 0.f);
    } else {
#pragma unroll
        for (int t = 0; t < 4; ++t)
#pragma unroll
            for (int r = 0; r < 8; ++r) acc[t][r] += bia[t];
    }
    if (Cf) {
#pragma unroll
        for (int t = 0; t < 4; ++t)
#pragma unroll
            for (int r = 0; r < 8; ++r)
                Cf[(size_t)(rowb + r) * Ndim + bn + t * 16 + l16] = acc[t][r];
    }
    if (Ch) {
#pragma unroll
        for (int t = 0; t < 4; ++t)
#pragma unroll
            for (int r = 0; r < 8; ++r)
                Ch[(size_t)(rowb + r) * Ndim + bn + t * 16 + l16] = (_Float16)acc[t][r];
    }
}

// ---------------- attention (faithful to the reference's batch-mixing bug) ----------------
// scores[s] = (Q_s @ K_s^T) * scale -> softmax over c -> ao = P @ V_s
__global__ __launch_bounds__(256)
void attn_kernel(const _Float16* __restrict__ Qh, const _Float16* __restrict__ Kh,
                 const _Float16* __restrict__ Vh, _Float16* __restrict__ AOh) {
    __shared__ __align__(16) float    sS[B_ * B_];  // 32x32 scores
    __shared__ __align__(16) _Float16 sP[B_ * B_];  // softmax probs (f16)

    const int s    = blockIdx.x;
    const int tid  = threadIdx.x;
    const int lane = tid & 31;
    const int wave = tid >> 5;
    const int half = lane >> 4;
    const int l16  = lane & 15;
    const size_t base = (size_t)s * B_ * E_;
    const float scale = 0.03608439182435161f;  // 1/sqrt(768)

    // ---- scores: M=32, N=32, K=768; 2x2 tiles on waves 0..3 ----
    if (wave < 4) {
        const int tm = wave >> 1, tn = wave & 1;
        v8f acc = {};
        for (int k0 = 0; k0 < E_; k0 += 32) {
            const _Float16* qp = Qh + base + (size_t)(tm * 16 + l16) * E_ + k0 + half * 16;
            const v16h a = cat8(ld8(qp), ld8(qp + 8));
            // B = K^T: fragment column n is K-row n -> contiguous along e
            const _Float16* kp = Kh + base + (size_t)(tn * 16 + l16) * E_ + k0 + half * 16;
            const v16h b = cat8(ld8(kp), ld8(kp + 8));
            acc = __builtin_amdgcn_wmma_f32_16x16x32_f16(false, a, false, b,
                                                         (short)0, acc, false, false);
        }
#pragma unroll
        for (int r = 0; r < 8; ++r)
            sS[(tm * 16 + r + 8 * half) * B_ + tn * 16 + l16] = acc[r] * scale;
    }
    __syncthreads();

    // ---- softmax over c (row-wise, 32 rows) ----
    if (tid < B_) {
        float mx = -1e30f;
        for (int c = 0; c < B_; ++c) mx = fmaxf(mx, sS[tid * B_ + c]);
        float sum = 0.f;
        for (int c = 0; c < B_; ++c) sum += __expf(sS[tid * B_ + c] - mx);
        const float inv = 1.f / sum;
        for (int c = 0; c < B_; ++c)
            sP[tid * B_ + c] = (_Float16)(__expf(sS[tid * B_ + c] - mx) * inv);
    }
    __syncthreads();

    // ---- ao = P @ V : M=32, N=768, K=32 (one WMMA per tile) ----
    const _Float16* p0 = &sP[l16 * B_ + half * 16];
    const _Float16* p1 = &sP[(16 + l16) * B_ + half * 16];
    const v16h a0 = cat8(ld8(p0), ld8(p0 + 8));
    const v16h a1 = cat8(ld8(p1), ld8(p1 + 8));
    for (int t = wave; t < 2 * (E_ / 16); t += 8) {
        const int tm = t & 1, tn = t >> 1;
        v16h b;
#pragma unroll
        for (int kk = 0; kk < 16; ++kk)
            b[kk] = Vh[base + (size_t)(half * 16 + kk) * E_ + tn * 16 + l16];
        v8f acc = {};
        acc = __builtin_amdgcn_wmma_f32_16x16x32_f16(false, tm ? a1 : a0, false, b,
                                                     (short)0, acc, false, false);
#pragma unroll
        for (int r = 0; r < 8; ++r)
            AOh[base + (size_t)(tm * 16 + r + 8 * half) * E_ + tn * 16 + l16] =
                (_Float16)acc[r];
    }
}

// ---------------- residual + layernorm; writes f32 state and f16 GEMM input ----------------
__global__ __launch_bounds__(256)
void add_ln_kernel(const float* __restrict__ X, const float* __restrict__ Y,
                   const float* __restrict__ g, const float* __restrict__ be,
                   float* __restrict__ Xo, _Float16* __restrict__ Xh) {
    __shared__ float s1[256], s2[256];
    const int t   = blockIdx.x;
    const int tid = threadIdx.x;
    float vals[3];
    float sum = 0.f, sq = 0.f;
#pragma unroll
    for (int i = 0; i < 3; ++i) {
        const int e = tid + i * 256;
        const float v = X[(size_t)t * E_ + e] + Y[(size_t)t * E_ + e];
        vals[i] = v; sum += v; sq += v * v;
    }
    s1[tid] = sum; s2[tid] = sq;
    __syncthreads();
    for (int st = 128; st > 0; st >>= 1) {
        if (tid < st) { s1[tid] += s1[tid + st]; s2[tid] += s2[tid + st]; }
        __syncthreads();
    }
    const float mean = s1[0] * (1.f / E_);
    const float var  = s2[0] * (1.f / E_) - mean * mean;
    const float rs   = rsqrtf(var + 1e-5f);
#pragma unroll
    for (int i = 0; i < 3; ++i) {
        const int e = tid + i * 256;
        const float o = (vals[i] - mean) * rs * g[e] + be[e];
        Xo[(size_t)t * E_ + e] = o;
        Xh[(size_t)t * E_ + e] = (_Float16)o;
    }
}

// ---------------- classifier head: out[b,c] = x[0,b,:] @ Wc + bc ----------------
__global__ __launch_bounds__(64)
void cls_kernel(const float* __restrict__ X, const float* __restrict__ Wc,
                const float* __restrict__ bc, float* __restrict__ out) {
    const int t = threadIdx.x;
    if (t >= B_ * C_) return;
    const int b = t >> 1, c = t & 1;
    float acc = bc[c];
    for (int e = 0; e < E_; ++e)
        acc += X[(size_t)b * E_ + e] * Wc[e * C_ + c];  // rows of x[s=0] are tokens 0..31
    out[t] = acc;
}

// ---------------- host-side orchestration ----------------
extern "C" void kernel_launch(void* const* d_in, const int* in_sizes, int n_in,
                              void* d_out, int out_size, void* d_ws, size_t ws_size,
                              hipStream_t stream) {
    (void)in_sizes; (void)n_in; (void)out_size; (void)ws_size;
    const int*   ids = (const int*)  d_in[0];
    const float* tok = (const float*)d_in[1];
    const float* pos = (const float*)d_in[2];
    const float* Wq  = (const float*)d_in[3];
    const float* bq  = (const float*)d_in[4];
    const float* Wk  = (const float*)d_in[5];
    const float* bk  = (const float*)d_in[6];
    const float* Wv  = (const float*)d_in[7];
    const float* bv  = (const float*)d_in[8];
    const float* Wo  = (const float*)d_in[9];
    const float* bo  = (const float*)d_in[10];
    const float* g1  = (const float*)d_in[11];
    const float* be1 = (const float*)d_in[12];
    const float* g2  = (const float*)d_in[13];
    const float* be2 = (const float*)d_in[14];
    const float* W1  = (const float*)d_in[15];
    const float* bf1 = (const float*)d_in[16];
    const float* W2  = (const float*)d_in[17];
    const float* bf2 = (const float*)d_in[18];
    const float* Wc  = (const float*)d_in[19];
    const float* bc  = (const float*)d_in[20];

    char* p = (char*)d_ws;
    size_t off = 0;
    auto alloc = [&](size_t bytes) -> void* {
        void* q = p + off;
        off += (bytes + 255) & ~(size_t)255;
        return q;
    };

    // transposed f16 weights: Wt[N,K]
    _Float16* wqT = (_Float16*)alloc((size_t)L_ * E_ * E_ * 2);
    _Float16* wkT = (_Float16*)alloc((size_t)L_ * E_ * E_ * 2);
    _Float16* wvT = (_Float16*)alloc((size_t)L_ * E_ * E_ * 2);
    _Float16* woT = (_Float16*)alloc((size_t)L_ * E_ * E_ * 2);
    _Float16* w1T = (_Float16*)alloc((size_t)L_ * E_ * F_ * 2);
    _Float16* w2T = (_Float16*)alloc((size_t)L_ * F_ * E_ * 2);
    float*    X   = (float*)   alloc((size_t)M_ * E_ * 4);
    _Float16* Xh  = (_Float16*)alloc((size_t)M_ * E_ * 2);
    _Float16* Qh  = (_Float16*)alloc((size_t)M_ * E_ * 2);
    _Float16* Kh  = (_Float16*)alloc((size_t)M_ * E_ * 2);
    _Float16* Vh  = (_Float16*)alloc((size_t)M_ * E_ * 2);
    _Float16* AOh = (_Float16*)alloc((size_t)M_ * E_ * 2);
    _Float16* Hh  = (_Float16*)alloc((size_t)M_ * F_ * 2);
    float*    Yf  = (float*)   alloc((size_t)M_ * E_ * 4);

    // transpose-convert weights once per call (tiny vs. the GEMM work)
    {
        dim3 gEE(E_ / 256, E_, L_);          // [E,E] -> [E,E]
        cvt_t_kernel<<<gEE, 256, 0, stream>>>(Wq, wqT, E_, E_);
        cvt_t_kernel<<<gEE, 256, 0, stream>>>(Wk, wkT, E_, E_);
        cvt_t_kernel<<<gEE, 256, 0, stream>>>(Wv, wvT, E_, E_);
        cvt_t_kernel<<<gEE, 256, 0, stream>>>(Wo, woT, E_, E_);
        dim3 gEF(E_ / 256, F_, L_);          // [E,F] -> [F,E]
        cvt_t_kernel<<<gEF, 256, 0, stream>>>(W1, w1T, E_, F_);
        dim3 gFE(F_ / 256, E_, L_);          // [F,E] -> [E,F]
        cvt_t_kernel<<<gFE, 256, 0, stream>>>(W2, w2T, F_, E_);
    }

    embed_kernel<<<M_, 256, 0, stream>>>(ids, tok, pos, X, Xh);

    const dim3 gE(E_ / TN, M_ / TM);   // (12, 128)
    const dim3 gF(F_ / TN, M_ / TM);   // (48, 128)

    for (int l = 0; l < L_; ++l) {
        const size_t wEE = (size_t)l * E_ * E_;
        const size_t wEF = (size_t)l * E_ * F_;
        gemm_f16_kernel<<<gE, 256, 0, stream>>>(Xh, wqT + wEE, bq + l * E_,
                                                nullptr, Qh, E_, E_, 0);
        gemm_f16_kernel<<<gE, 256, 0, stream>>>(Xh, wkT + wEE, bk + l * E_,
                                                nullptr, Kh, E_, E_, 0);
        gemm_f16_kernel<<<gE, 256, 0, stream>>>(Xh, wvT + wEE, bv + l * E_,
                                                nullptr, Vh, E_, E_, 0);
        attn_kernel<<<S_, 256, 0, stream>>>(Qh, Kh, Vh, AOh);
        gemm_f16_kernel<<<gE, 256, 0, stream>>>(AOh, woT + wEE, bo + l * E_,
                                                Yf, nullptr, E_, E_, 0);
        add_ln_kernel<<<M_, 256, 0, stream>>>(X, Yf, g1 + l * E_, be1 + l * E_, X, Xh);
        gemm_f16_kernel<<<gF, 256, 0, stream>>>(Xh, w1T + wEF, bf1 + l * F_,
                                                nullptr, Hh, F_, E_, 1);
        gemm_f16_kernel<<<gE, 256, 0, stream>>>(Hh, w2T + wEF, bf2 + l * E_,
                                                Yf, nullptr, E_, F_, 0);
        add_ln_kernel<<<M_, 256, 0, stream>>>(X, Yf, g2 + l * E_, be2 + l * E_, X, Xh);
    }

    cls_kernel<<<1, 64, 0, stream>>>(X, Wc, bc, (float*)d_out);
}
